// RGHAT_52226802319806
// MI455X (gfx1250) — matrix operand order repaired and checked by reading, not compiled, not run on hardware
//
#include <hip/hip_runtime.h>

typedef __bf16 bf16;
typedef __attribute__((ext_vector_type(16))) __bf16 v16bf;
typedef __attribute__((ext_vector_type(8)))  float  v8f;

#define N_ENT   20000
#define N_RELS  64
#define N_EDGE  80000
#define DIM     768
#define BATCH   512
#define NEGK    32
#define CFLAT   44160

__device__ __forceinline__ float leaky_f(float x, float s) { return x >= 0.f ? x : s * x; }

__device__ __forceinline__ unsigned short f2bf(float f) {
    bf16 t = (bf16)f;
    return __builtin_bit_cast(unsigned short, t);
}
__device__ __forceinline__ float bf2f(unsigned short u) {
    return (float)__builtin_bit_cast(bf16, u);
}

// monotonic float <-> uint mapping for atomicMax on floats
__device__ __forceinline__ unsigned f2key(float f) {
    unsigned b = __float_as_uint(f);
    return (b & 0x80000000u) ? ~b : (b | 0x80000000u);
}
__device__ __forceinline__ float key2f(unsigned k) {
    unsigned b = (k & 0x80000000u) ? (k & 0x7FFFFFFFu) : ~k;
    return __uint_as_float(b);
}

// ---------------------------------------------------------------------------
// Generic WMMA GEMM: C[m, n] = sum_k A(m,k) * W[n_global, k]   (N total = 768)
// Block tile 128x64, TILE_K = 64, double-buffered LDS, software pipelined:
//   round i: issue global loads for tile i+1, run 2x(4 WMMA) on tile i from
//   LDS (overlapping the loads), then cvt+store tile i+1, barrier.
// 8 waves (4x2 in MxN), each wave owns a 32x32 patch (2x2 WMMA 16x16x32 bf16).
// AMODE: A-element source (gathers / K-concat / fused elementwise).
// OMODE: epilogue.
// ---------------------------------------------------------------------------
template <int AMODE, int OMODE>
__global__ __launch_bounds__(256) void gemm_wmma(
    const float* __restrict__ a0, const float* __restrict__ a1,
    const bf16* __restrict__ abf, const int* __restrict__ idx0,
    const int* __restrict__ idx1, const float* __restrict__ Bw,
    int M, int K, float* __restrict__ outF, bf16* __restrict__ outBf)
{
    // stride 66 ushort = 132B = 33 banks; gcd(33,64)=1 -> conflict-free cols
    __shared__ unsigned short As[2][128][66];
    __shared__ unsigned short Bs[2][64][66];

    const int tid  = threadIdx.x;
    const int lane = tid & 31;
    const int wave = tid >> 5;
    const int wm   = wave >> 1;          // 0..3  (M direction)
    const int wn   = wave & 1;           // 0..1  (N direction)
    const int m0   = blockIdx.x * 128;
    const int n0   = blockIdx.y * 64;
    const int hf   = lane >> 4;
    const int l16  = lane & 15;

    v8f acc[2][2];
#pragma unroll
    for (int i = 0; i < 2; ++i)
#pragma unroll
        for (int j = 0; j < 2; ++j)
#pragma unroll
            for (int r = 0; r < 8; ++r) acc[i][j][r] = 0.f;

    // staging coords: 16 threads cover 64 K-values (4 each); 16 row groups
    const int tkc = (tid & 15) * 4;
    const int trc = tid >> 4;            // 0..15

    // per-thread A row indices (8 rows: trc + 16*g)
    int  ga[8], gb[8];
    bool valid[8];
#pragma unroll
    for (int g = 0; g < 8; ++g) {
        int grow = m0 + trc + g * 16;
        valid[g] = grow < M;
        int gg   = valid[g] ? grow : 0;
        if (AMODE == 0)      { ga[g] = idx0[gg]; gb[g] = idx1[gg]; }
        else if (AMODE == 1) { ga[g] = gg;       gb[g] = idx1[gg]; }
        else                 { ga[g] = gg;       gb[g] = gg;       }
    }

    float4  aF[8];   // f32 staging regs (AMODE 0/2/3, AMODE1 tail)
    ushort4 aH[8];   // bf16 staging regs (AMODE1 head, AMODE4)
    float4  bF[4];

    // ---- issue global loads for one 64-wide K slab into registers ----
    auto loadA = [&](int k0) {
#pragma unroll
        for (int g = 0; g < 8; ++g) {
            if (AMODE == 1 && k0 < 768) {
                aH[g] = valid[g]
                      ? *(const ushort4*)((const unsigned short*)abf + (size_t)ga[g] * 768 + k0 + tkc)
                      : make_ushort4(0, 0, 0, 0);
            } else if (AMODE == 4) {
                aH[g] = valid[g]
                      ? *(const ushort4*)((const unsigned short*)abf + (size_t)ga[g] * K + k0 + tkc)
                      : make_ushort4(0, 0, 0, 0);
            } else {
                float4 fv = make_float4(0.f, 0.f, 0.f, 0.f);
                if (valid[g]) {
                    if (AMODE == 0) {
                        fv = (k0 < 768)
                           ? *(const float4*)(a0 + (size_t)ga[g] * 768 + k0 + tkc)
                           : *(const float4*)(a1 + (size_t)gb[g] * 768 + (k0 - 768) + tkc);
                    } else if (AMODE == 1) {   // k0 >= 768: tail entity gather
                        fv = *(const float4*)(a0 + (size_t)gb[g] * 768 + (k0 - 768) + tkc);
                    } else {                   // AMODE 2/3: ent (+|*) side, fused
                        float4 x = *(const float4*)(a0 + (size_t)ga[g] * 768 + k0 + tkc);
                        float4 y = *(const float4*)(a1 + (size_t)ga[g] * 768 + k0 + tkc);
                        if (AMODE == 2) { fv.x = x.x + y.x; fv.y = x.y + y.y; fv.z = x.z + y.z; fv.w = x.w + y.w; }
                        else            { fv.x = x.x * y.x; fv.y = x.y * y.y; fv.z = x.z * y.z; fv.w = x.w * y.w; }
                    }
                }
                aF[g] = fv;
            }
        }
    };
    auto loadB = [&](int k0) {
#pragma unroll
        for (int g = 0; g < 4; ++g) {
            int n = trc + g * 16;
            bF[g] = *(const float4*)(Bw + (size_t)(n0 + n) * K + k0 + tkc);
        }
    };
    // ---- convert + store staged regs into LDS buffer ----
    auto storeAB = [&](int buf, int k0) {
#pragma unroll
        for (int g = 0; g < 8; ++g) {
            int row = trc + g * 16;
            unsigned short* d = &As[buf][row][tkc];
            if ((AMODE == 1 && k0 < 768) || AMODE == 4) {
                d[0] = aH[g].x; d[1] = aH[g].y; d[2] = aH[g].z; d[3] = aH[g].w;
            } else {
                d[0] = f2bf(aF[g].x); d[1] = f2bf(aF[g].y);
                d[2] = f2bf(aF[g].z); d[3] = f2bf(aF[g].w);
            }
        }
#pragma unroll
        for (int g = 0; g < 4; ++g) {
            int n = trc + g * 16;
            unsigned short* d = &Bs[buf][n][tkc];
            d[0] = f2bf(bF[g].x); d[1] = f2bf(bF[g].y);
            d[2] = f2bf(bF[g].z); d[3] = f2bf(bF[g].w);
        }
    };
    // ---- consume one LDS buffer: 2 K-substeps x (2x2) WMMA ----
    auto compute = [&](int buf) {
#pragma unroll
        for (int ks = 0; ks < 2; ++ks) {
            union { v16bf v; unsigned u[8]; } af[2], bfx[2];
#pragma unroll
            for (int i = 0; i < 2; ++i) {
                int mrow = wm * 32 + i * 16 + l16;
#pragma unroll
                for (int vv = 0; vv < 8; ++vv) {
                    int kk = ks * 32 + 8 * hf + ((vv < 4) ? 2 * vv : 8 + 2 * vv); // A 16x32 map
                    af[i].u[vv] = *(const unsigned*)&As[buf][mrow][kk];
                }
            }
#pragma unroll
            for (int j = 0; j < 2; ++j) {
                int ncol = wn * 32 + j * 16 + l16;
#pragma unroll
                for (int vv = 0; vv < 8; ++vv) {
                    int kk = ks * 32 + 8 * hf + ((vv < 4) ? 2 * vv : 8 + 2 * vv); // B 32x16 map
                    bfx[j].u[vv] = *(const unsigned*)&Bs[buf][ncol][kk];
                }
            }
#pragma unroll
            for (int i = 0; i < 2; ++i)
#pragma unroll
                for (int j = 0; j < 2; ++j)
                    acc[i][j] = __builtin_amdgcn_wmma_f32_16x16x32_bf16(
                        false, af[i].v, false, bfx[j].v, (short)0, acc[i][j],
                        false, false);
        }
    };

    // ---- pipelined main loop ----
    int buf = 0;
    loadA(0);
    loadB(0);
    storeAB(0, 0);
    __syncthreads();
    for (int k0 = 64; k0 < K; k0 += 64) {
        loadA(k0);              // global loads for next slab (async wrt WMMA)
        loadB(k0);
        compute(buf);           // WMMA on current buffer while loads fly
        storeAB(buf ^ 1, k0);   // waits loads, fills other buffer
        __syncthreads();
        buf ^= 1;
    }
    compute(buf);

    // ---- epilogue (C/D layout: VGPR r -> M = r + 8*half, N = lane&15) ----
#pragma unroll
    for (int i = 0; i < 2; ++i) {
#pragma unroll
        for (int j = 0; j < 2; ++j) {
            int colg = n0 + wn * 32 + j * 16 + l16;
#pragma unroll
            for (int r = 0; r < 8; ++r) {
                int rowg = m0 + wm * 32 + i * 16 + hf * 8 + r;
                if (rowg < M) {
                    float  val = acc[i][j][r];
                    size_t o   = (size_t)rowg * DIM + colg;
                    if (OMODE == 0)      outBf[o]  = (bf16)val;
                    else if (OMODE == 1) outF[o]   = leaky_f(val, 0.01f);
                    else if (OMODE == 2) outF[o]  += leaky_f(val, 0.01f);
                    else                 outF[o]   = val > 0.f ? val : 0.f;
                }
            }
        }
    }
}

// ---------------------------------------------------------------------------
// init: zero side, seg-max keys, seg-sums
// ---------------------------------------------------------------------------
__global__ void init_kernel(float* side, unsigned* k1, unsigned* k2,
                            float* s1, float* s2)
{
    size_t j = (size_t)blockIdx.x * 256 + threadIdx.x;
    if (j < (size_t)N_ENT * DIM) side[j] = 0.f;
    if (j < N_ENT) { k1[j] = 0u; k2[j] = 0u; s1[j] = 0.f; s2[j] = 0.f; }
}

// ---------------------------------------------------------------------------
// score = leaky_0.2( row(bf16) . vec ), one wave per row
// ---------------------------------------------------------------------------
__global__ __launch_bounds__(256) void score_kernel(
    const bf16* __restrict__ a, const float* __restrict__ vec,
    float* __restrict__ out, int E)
{
    int w    = (int)((blockIdx.x * blockDim.x + threadIdx.x) >> 5);
    int lane = threadIdx.x & 31;
    if (w >= E) return;
    float s = 0.f;
    for (int k = lane; k < DIM; k += 32)
        s += (float)a[(size_t)w * DIM + k] * vec[k];
#pragma unroll
    for (int off = 16; off; off >>= 1) s += __shfl_down(s, off, 32);
    if (lane == 0) out[w] = leaky_f(s, 0.2f);
}

__global__ void segmax_kernel(const float* s1, const float* s2,
                              const int* segh, unsigned* k1, unsigned* k2, int E)
{
    int i = blockIdx.x * blockDim.x + threadIdx.x;
    if (i >= E) return;
    int h = segh[i];
    atomicMax(&k1[h], f2key(s1[i]));
    atomicMax(&k2[h], f2key(s2[i]));
}

__global__ void expsum_kernel(const float* s1, const float* s2,
                              const unsigned* k1, const unsigned* k2,
                              const int* segh, float* e1, float* e2,
                              float* sum1, float* sum2, int E)
{
    int i = blockIdx.x * blockDim.x + threadIdx.x;
    if (i >= E) return;
    int   h = segh[i];
    float a = __expf(s1[i] - key2f(k1[h]));
    float b = __expf(s2[i] - key2f(k2[h]));
    e1[i] = a; e2[i] = b;
    atomicAdd(&sum1[h], a);
    atomicAdd(&sum2[h], b);
}

// side[h] += alpha_e * b_hrt[e],  one block per edge
__global__ __launch_bounds__(256) void side_kernel(
    const float* e1, const float* e2, const float* sum1, const float* sum2,
    const int* segh, const bf16* __restrict__ bbf, float* __restrict__ side)
{
    int e = blockIdx.x;
    int h = segh[e];
    float alpha = (e1[e] / (sum1[h] + 1e-24f)) * (e2[e] / (sum2[h] + 1e-24f));
    for (int j = threadIdx.x; j < DIM; j += 256)
        atomicAdd(&side[(size_t)h * DIM + j], alpha * (float)bbf[(size_t)e * DIM + j]);
}

// ---------------------------------------------------------------------------
// ConvE conv: image (48x32) in LDS, 3x3 VALID, 32 ch, relu, flatten -> bf16
// one block per batch element
// ---------------------------------------------------------------------------
__global__ __launch_bounds__(256) void conv_kernel(
    const float* __restrict__ enc, const float* __restrict__ rel,
    const float* __restrict__ cw, const int* hI, const int* rI,
    bf16* __restrict__ xconv)
{
    __shared__ float img[48][32];
    __shared__ float w[32][3][3];
    int b = blockIdx.x, t = threadIdx.x;
    int hrow = hI[b], rrow = rI[b];
    for (int i = t; i < DIM; i += 256) {
        img[i >> 5][i & 31]        = enc[(size_t)hrow * DIM + i];
        img[24 + (i >> 5)][i & 31] = rel[(size_t)rrow * DIM + i];
    }
    for (int i = t; i < 288; i += 256) ((float*)w)[i] = cw[i];
    __syncthreads();
    for (int idx = t; idx < CFLAT; idx += 256) {
        int c   = idx / 1380;
        int rem = idx - c * 1380;
        int i   = rem / 30;
        int j   = rem - i * 30;
        float acc = 0.f;
#pragma unroll
        for (int di = 0; di < 3; ++di)
#pragma unroll
            for (int dj = 0; dj < 3; ++dj)
                acc += w[c][di][dj] * img[i + di][j + dj];
        xconv[(size_t)b * CFLAT + idx] = (bf16)(acc > 0.f ? acc : 0.f);
    }
}

// sigmoid( x[b] . enc[t] ), one wave per (b, candidate)
__global__ __launch_bounds__(256) void final_kernel(
    const float* __restrict__ x, const float* __restrict__ enc,
    const int* pos_t, const int* neg_t, float* __restrict__ out)
{
    int gw   = (int)((blockIdx.x * blockDim.x + threadIdx.x) >> 5);
    int lane = threadIdx.x & 31;
    if (gw >= BATCH * (NEGK + 1)) return;
    int b = gw / (NEGK + 1);
    int s = gw - b * (NEGK + 1);
    int t = (s == 0) ? pos_t[b] : neg_t[b * NEGK + (s - 1)];
    float acc = 0.f;
    for (int k = lane; k < DIM; k += 32)
        acc += x[(size_t)b * DIM + k] * enc[(size_t)t * DIM + k];
#pragma unroll
    for (int off = 16; off; off >>= 1) acc += __shfl_down(acc, off, 32);
    if (lane == 0) {
        float sg = 1.f / (1.f + __expf(-acc));
        if (s == 0) out[b] = sg;
        else        out[BATCH + b * NEGK + (s - 1)] = sg;
    }
}

// ---------------------------------------------------------------------------
extern "C" void kernel_launch(void* const* d_in, const int* in_sizes, int n_in,
                              void* d_out, int out_size, void* d_ws, size_t ws_size,
                              hipStream_t stream)
{
    const float* ent   = (const float*)d_in[0];
    const float* rel   = (const float*)d_in[1];
    const float* W1    = (const float*)d_in[2];
    const float* W2    = (const float*)d_in[3];
    const float* W3    = (const float*)d_in[4];
    const float* W4    = (const float*)d_in[5];
    const float* Q     = (const float*)d_in[6];
    const float* p     = (const float*)d_in[7];
    const float* q     = (const float*)d_in[8];
    const float* convw = (const float*)d_in[9];
    const int*   eh    = (const int*)d_in[10];
    const int*   er    = (const int*)d_in[11];
    const int*   et    = (const int*)d_in[12];
    const int*   hI    = (const int*)d_in[13];
    const int*   rI    = (const int*)d_in[14];
    const int*   post  = (const int*)d_in[15];
    const int*   negt  = (const int*)d_in[16];
    float*       out   = (float*)d_out;
    (void)in_sizes; (void)n_in; (void)out_size; (void)ws_size;

    char*  ws  = (char*)d_ws;
    size_t off = 0;
    auto carve = [&](size_t bytes) {
        char* ptr = ws + off;
        off = (off + bytes + 255) & ~(size_t)255;
        return ptr;
    };
    bf16*     a_hr  = (bf16*)    carve((size_t)N_EDGE * DIM * 2);
    bf16*     b_hrt = (bf16*)    carve((size_t)N_EDGE * DIM * 2);
    float*    hr_s  = (float*)   carve((size_t)N_EDGE * 4);
    float*    hrt_s = (float*)   carve((size_t)N_EDGE * 4);
    float*    e1    = (float*)   carve((size_t)N_EDGE * 4);
    float*    e2    = (float*)   carve((size_t)N_EDGE * 4);
    unsigned* k1    = (unsigned*)carve((size_t)N_ENT * 4);
    unsigned* k2    = (unsigned*)carve((size_t)N_ENT * 4);
    float*    s1    = (float*)   carve((size_t)N_ENT * 4);
    float*    s2    = (float*)   carve((size_t)N_ENT * 4);
    float*    side  = (float*)   carve((size_t)N_ENT * DIM * 4);
    float*    enc   = (float*)   carve((size_t)N_ENT * DIM * 4);
    bf16*     xconv = (bf16*)    carve((size_t)BATCH * CFLAT * 2);
    float*    xout  = (float*)   carve((size_t)BATCH * DIM * 4);

    // 1) zero accumulators
    init_kernel<<<(N_ENT * DIM + 255) / 256, 256, 0, stream>>>(side, k1, k2, s1, s2);

    // 2) a_hr = concat(h_e, r_e) @ W1.T   (E x 768, bf16)
    gemm_wmma<0, 0><<<dim3(N_EDGE / 128, DIM / 64), 256, 0, stream>>>(
        ent, rel, nullptr, eh, er, W1, N_EDGE, 2 * DIM, nullptr, a_hr);
    // 3) hr_score = leaky(a_hr . p, 0.2)
    score_kernel<<<N_EDGE / 8, 256, 0, stream>>>(a_hr, p, hr_s, N_EDGE);
    // 4) b_hrt = concat(a_hr, t_e) @ W2.T
    gemm_wmma<1, 0><<<dim3(N_EDGE / 128, DIM / 64), 256, 0, stream>>>(
        ent, nullptr, a_hr, nullptr, et, W2, N_EDGE, 2 * DIM, nullptr, b_hrt);
    // 5) hrt_score = leaky(b_hrt . q, 0.2)
    score_kernel<<<N_EDGE / 8, 256, 0, stream>>>(b_hrt, q, hrt_s, N_EDGE);
    // 6-8) segment softmax (max, exp+sum) and weighted scatter into side
    segmax_kernel<<<(N_EDGE + 255) / 256, 256, 0, stream>>>(hr_s, hrt_s, eh, k1, k2, N_EDGE);
    expsum_kernel<<<(N_EDGE + 255) / 256, 256, 0, stream>>>(hr_s, hrt_s, k1, k2, eh,
                                                            e1, e2, s1, s2, N_EDGE);
    side_kernel<<<N_EDGE, 256, 0, stream>>>(e1, e2, s1, s2, eh, b_hrt, side);
    // 9) enc = leaky((ent+side)@W3.T, .01) + leaky((ent*side)@W4.T, .01)
    gemm_wmma<2, 1><<<dim3((N_ENT + 127) / 128, DIM / 64), 256, 0, stream>>>(
        ent, side, nullptr, nullptr, nullptr, W3, N_ENT, DIM, enc, nullptr);
    gemm_wmma<3, 2><<<dim3((N_ENT + 127) / 128, DIM / 64), 256, 0, stream>>>(
        ent, side, nullptr, nullptr, nullptr, W4, N_ENT, DIM, enc, nullptr);
    // 10) ConvE conv + relu + flatten (bf16)
    conv_kernel<<<BATCH, 256, 0, stream>>>(enc, rel, convw, hI, rI, xconv);
    // 11) x = relu(xconv @ Q.T)
    gemm_wmma<4, 3><<<dim3(BATCH / 128, DIM / 64), 256, 0, stream>>>(
        nullptr, nullptr, xconv, nullptr, nullptr, Q, BATCH, CFLAT, xout, nullptr);
    // 12) pos/neg sigmoid scores
    final_kernel<<<(BATCH * (NEGK + 1)) / 8, 256, 0, stream>>>(xout, enc, post, negt, out);
}
